// AFNO1D_46248207843958
// MI455X (gfx1250) — matrix operand order repaired and checked by reading, not compiled, not run on hardware
//
#include <hip/hip_runtime.h>
#include <hip/hip_bf16.h>

typedef _Float16 h16;
typedef __attribute__((ext_vector_type(16))) _Float16 v16h;
typedef __attribute__((ext_vector_type(8)))  _Float16 v8h;
typedef __attribute__((ext_vector_type(4)))  _Float16 v4h;
typedef __attribute__((ext_vector_type(8)))  float    v8f;
typedef __attribute__((ext_vector_type(4)))  float    v4f;

#define HIDDEN 1024
#define LAMBDA 0.01f
#define INV_SQRT_N 0.03125f        // 1/sqrt(1024)
#define TWO_PI 6.28318530717958647692f

// LDS row strides (f16 elems), multiples of 8 so all v8h accesses stay 16B aligned.
// Stride % 64-bank pattern gives conflict-free b128 row-parallel reads.
#define SA_STR  1032
#define SO_STR  2056
#define MTILE   32                 // token rows per workgroup (2 WMMA row-groups)

// ---------------- fragment helpers (CDNA5 WMMA f16 16x16x32) ----------------
// A (16x32, MxK): lane<16 -> row M=lane, slots 0..7 = K 0..7, slots 8..15 = K 16..23
//                 lane>=16 -> row M=lane-16, K 8..15 and 24..31  (ISA 7.12.2)
__device__ __forceinline__ v16h load_frag_A(const h16* base, int rowStride,
                                            int kbase, int lane) {
    const int r = lane & 15, h = lane >> 4;
    const h16* p = base + r * rowStride + kbase + 8 * h;
    v8h lo = *(const v8h*)(p);
    v8h hi = *(const v8h*)(p + 16);
    return __builtin_shufflevector(lo, hi, 0,1,2,3,4,5,6,7,8,9,10,11,12,13,14,15);
}

// B (32x16, KxN): lane&15 = column N, slots s -> K = (s&7) + 8*(lane>>4) + 16*(s>>3)
// (same K order as load_frag_A: lanes0-15 K=0-7,16-23; lanes16-31 K=8-15,24-31).
// Prep kernels store fragments so lane's 16 f16 are contiguous at lane*16.
__device__ __forceinline__ v16h load_frag_B(const h16* fragbase, int lane) {
    const h16* p = fragbase + lane * 16;
    v8h lo = *(const v8h*)(p);
    v8h hi = *(const v8h*)(p + 8);
    return __builtin_shufflevector(lo, hi, 0,1,2,3,4,5,6,7,8,9,10,11,12,13,14,15);
}

__device__ __forceinline__ v8f wmma_f16(v16h a, v16h b, v8f c) {
    return __builtin_amdgcn_wmma_f32_16x16x32_f16(false, a, false, b,
                                                  (short)0, c, false, false);
}

// B-fragment element coordinates from (lane, slot)
__device__ __forceinline__ void bfrag_coord(int lane, int s, int& kk, int& nn) {
    kk = (s & 7) + 8 * (lane >> 4) + 16 * (s >> 3);
    nn = lane & 15;
}

// ---------------- prep 1: folded FFT+layer1 matrix  M1[1024 x 2048] ----------------
// M1r[c][j] = (1/32) sum_i cos(phi)*w1r + sin(phi)*w1i      (part 0)
// M1i[c][j] = (1/32) sum_i -sin(phi)*w1r + cos(phi)*w1i     (part 1)
// phi = 2*pi*(k*128+i)*c/1024 ; columns j = k*256 + part*128 + o
__global__ void __launch_bounds__(256)
afno_prep_m1(const float* __restrict__ w1, h16* __restrict__ M1) {
    int e = blockIdx.x * 256 + threadIdx.x;           // 2,097,152 elements
    int frag = e >> 9, within = e & 511;
    int lane = within >> 4, s = within & 15;
    int kt = frag >> 7, nt = frag & 127;              // Ktiles=32, Ntiles=128
    int kk, nn; bfrag_coord(lane, s, kk, nn);
    int c = kt * 32 + kk;                             // input channel (K dim)
    int j = nt * 16 + nn;                             // output column (N dim)
    int k = j >> 8, part = (j >> 7) & 1, o = j & 127;

    // phi_i = phi0 + i*delta ; rotate instead of 128 sincos calls
    float phi0 = (float)((k * c) & 7) * (TWO_PI / 8.0f);
    float delta = (float)c * (TWO_PI / 1024.0f);
    float cs, sn, cd, sd;
    __sincosf(phi0, &sn, &cs);
    __sincosf(delta, &sd, &cd);

    const float* w1r = w1 + k * 16384 + o;            // w1[0][k][i][o]
    const float* w1i = w1 + 131072 + k * 16384 + o;   // w1[1][k][i][o]
    float acc = 0.f;
    for (int i = 0; i < 128; ++i) {
        float wr = w1r[i * 128], wi = w1i[i * 128];
        acc += (part == 0) ? (cs * wr + sn * wi) : (cs * wi - sn * wr);
        float ncs = cs * cd - sn * sd;
        float nsn = sn * cd + cs * sd;
        cs = ncs; sn = nsn;
    }
    M1[e] = (h16)(acc * INV_SQRT_N);
}

// ---------------- prep 2: combined complex layer-2 matrix  W2c[8][256 x 256] -------
// [o2r|o2i] = [o1r|o1i] * [[W2r, W2i], [-W2i, W2r]]
__global__ void __launch_bounds__(256)
afno_prep_w2(const float* __restrict__ w2, h16* __restrict__ W2c) {
    int e = blockIdx.x * 256 + threadIdx.x;           // 524,288 elements
    int blk = e >> 16, r = e & 65535;
    int frag = r >> 9, within = r & 511;
    int lane = within >> 4, s = within & 15;
    int kt = frag >> 4, nt = frag & 15;               // Ktiles=8, Ntiles=16
    int kk, nn; bfrag_coord(lane, s, kk, nn);
    int K = kt * 32 + kk;                             // 0..255
    int N = nt * 16 + nn;                             // 0..255
    int pin = K >> 7, i = K & 127;
    int pout = N >> 7, o = N & 127;
    const float* w2r = w2 + blk * 16384;              // w2[0][blk]
    const float* w2i = w2 + 131072 + blk * 16384;     // w2[1][blk]
    float v;
    if (pout == 0) v = (pin == 0) ?  w2r[i * 128 + o] : -w2i[i * 128 + o];
    else           v = (pin == 0) ?  w2i[i * 128 + o] :  w2r[i * 128 + o];
    W2c[e] = (h16)v;
}

// ---------------- prep 3: real-IFFT matrix  Cm[2048 x 1024] ----------------
// out[n] = (1/32) sum_f ( o2r[f]*cos(2 pi n f/1024) - o2i[f]*sin(...) )
__global__ void __launch_bounds__(256)
afno_prep_cm(h16* __restrict__ Cm) {
    int e = blockIdx.x * 256 + threadIdx.x;           // 2,097,152 elements
    int frag = e >> 9, within = e & 511;
    int lane = within >> 4, s = within & 15;
    int kt = frag >> 6, nt = frag & 63;               // Ktiles=64, Ntiles=64
    int kk, nn; bfrag_coord(lane, s, kk, nn);
    int K = kt * 32 + kk;                             // 0..2047 (o2 column)
    int N = nt * 16 + nn;                             // 0..1023 (output channel)
    int k = K >> 8, part = (K >> 7) & 1, o = K & 127;
    int f = k * 128 + o;
    int m = (f * N) & 1023;                           // exact integer phase reduction
    float ang = (float)m * (TWO_PI / 1024.0f);
    float sv, cv;
    __sincosf(ang, &sv, &cv);
    Cm[e] = (h16)(((part == 0) ? cv : -sv) * INV_SQRT_N);
}

// ---------------- fused main kernel: 32-token tile, 3 chained WMMA GEMMs ----------
// LDS (263 KB, CDNA5 allows 320 KB/WG):
//   region0: sO1 [32 x SO_STR]
//   region1: time-shared: sA [32 x SA_STR] (load + stage1) -> sO2 [32 x SO_STR]
__global__ void __launch_bounds__(256)
afno_main(const float* __restrict__ x,
          const float* __restrict__ b1, const float* __restrict__ b2,
          const h16* __restrict__ M1, const h16* __restrict__ W2c,
          const h16* __restrict__ Cm, float* __restrict__ out) {
    extern __shared__ __align__(16) char smem_raw[];
    h16* sO1 = (h16*)smem_raw;                       // 32 x SO_STR
    h16* sA  = sO1 + MTILE * SO_STR;                 // 32 x SA_STR (aliased below)
    h16* sO2 = sA;                                   // 32 x SO_STR (after stage 1)

    const int tid  = threadIdx.x;
    const int lane = tid & 31, wave = tid >> 5;
    const int h = lane >> 4, n = lane & 15;
    const long tile = (long)blockIdx.x * MTILE;

    // load X tile (32x1024 f32) -> f16 LDS, float4-vectorized, coalesced
    for (int idx = tid; idx < MTILE * 256; idx += 256) {
        int r = idx >> 8, c4 = idx & 255;
        v4f xin = *(const v4f*)(x + (tile + r) * HIDDEN + c4 * 4);
        v4h xh = { (h16)xin.x, (h16)xin.y, (h16)xin.z, (h16)xin.w };
        *(v4h*)(sA + r * SA_STR + c4 * 4) = xh;
    }
    __syncthreads();

    // ---- Stage 1: pre1 = X * M1 ; +b1 ; relu  (K=1024 -> N=2048) ----
    for (int nt = wave; nt < 128; nt += 8) {
        v8f acc0 = {}, acc1 = {};
        #pragma unroll 4
        for (int kt = 0; kt < 32; ++kt) {
            const h16* bptr = M1 + (size_t)(kt * 128 + nt) * 512;
            int ktn = kt < 31 ? kt + 1 : 31;         // branchless prefetch clamp
            __builtin_prefetch(M1 + (size_t)(ktn * 128 + nt) * 512, 0, 1);
            v16h b  = load_frag_B(bptr, lane);
            v16h a0 = load_frag_A(sA,                SA_STR, kt * 32, lane);
            v16h a1 = load_frag_A(sA + 16 * SA_STR,  SA_STR, kt * 32, lane);
            acc0 = wmma_f16(a0, b, acc0);
            acc1 = wmma_f16(a1, b, acc1);
        }
        int j = nt * 16 + n;
        float bias = b1[((j >> 7) & 1) * 1024 + (j >> 8) * 128 + (j & 127)];
        #pragma unroll
        for (int v = 0; v < 8; ++v) {
            float v0 = acc0[v] + bias;
            float v1 = acc1[v] + bias;
            v0 = v0 > 0.f ? v0 : 0.f;
            v1 = v1 > 0.f ? v1 : 0.f;
            sO1[(8 * h + v) * SO_STR + j]        = (h16)v0;
            sO1[(16 + 8 * h + v) * SO_STR + j]   = (h16)v1;
        }
    }
    __syncthreads();   // stage-1 done: sA dead, region1 becomes sO2

    // ---- Stage 2: per-block complex layer 2 ; +b2 ; softshrink ----
    {
        const int k = wave;                  // 8 waves <-> 8 blocks
        for (int nt2 = 0; nt2 < 16; ++nt2) {
            v8f acc0 = {}, acc1 = {};
            #pragma unroll
            for (int kt2 = 0; kt2 < 8; ++kt2) {
                v16h b  = load_frag_B(W2c + (size_t)(k * 128 + kt2 * 16 + nt2) * 512, lane);
                v16h a0 = load_frag_A(sO1,                 SO_STR, k * 256 + kt2 * 32, lane);
                v16h a1 = load_frag_A(sO1 + 16 * SO_STR,   SO_STR, k * 256 + kt2 * 32, lane);
                acc0 = wmma_f16(a0, b, acc0);
                acc1 = wmma_f16(a1, b, acc1);
            }
            int jj = nt2 * 16 + n;
            float bias = b2[(jj >> 7) * 1024 + k * 128 + (jj & 127)];
            #pragma unroll
            for (int v = 0; v < 8; ++v) {
                float v0 = acc0[v] + bias;
                float v1 = acc1[v] + bias;
                v0 = (v0 > LAMBDA) ? (v0 - LAMBDA) : ((v0 < -LAMBDA) ? (v0 + LAMBDA) : 0.f);
                v1 = (v1 > LAMBDA) ? (v1 - LAMBDA) : ((v1 < -LAMBDA) ? (v1 + LAMBDA) : 0.f);
                sO2[(8 * h + v) * SO_STR + k * 256 + jj]      = (h16)v0;
                sO2[(16 + 8 * h + v) * SO_STR + k * 256 + jj] = (h16)v1;
            }
        }
    }
    __syncthreads();

    // ---- Stage 3: out = Re(ifft(o2)) + residual  (K=2048 -> N=1024) ----
    for (int nt3 = wave; nt3 < 64; nt3 += 8) {
        v8f acc0 = {}, acc1 = {};
        #pragma unroll 4
        for (int kt3 = 0; kt3 < 64; ++kt3) {
            const h16* bptr = Cm + (size_t)(kt3 * 64 + nt3) * 512;
            int ktn = kt3 < 63 ? kt3 + 1 : 63;       // branchless prefetch clamp
            __builtin_prefetch(Cm + (size_t)(ktn * 64 + nt3) * 512, 0, 1);
            v16h b  = load_frag_B(bptr, lane);
            v16h a0 = load_frag_A(sO2,               SO_STR, kt3 * 32, lane);
            v16h a1 = load_frag_A(sO2 + 16 * SO_STR, SO_STR, kt3 * 32, lane);
            acc0 = wmma_f16(a0, b, acc0);
            acc1 = wmma_f16(a1, b, acc1);
        }
        int ncol = nt3 * 16 + n;
        #pragma unroll
        for (int v = 0; v < 8; ++v) {
            long r0 = tile + 8 * h + v;
            long r1 = tile + 16 + 8 * h + v;
            out[r0 * HIDDEN + ncol] = acc0[v] + x[r0 * HIDDEN + ncol];
            out[r1 * HIDDEN + ncol] = acc1[v] + x[r1 * HIDDEN + ncol];
        }
    }
}

extern "C" void kernel_launch(void* const* d_in, const int* in_sizes, int n_in,
                              void* d_out, int out_size, void* d_ws, size_t ws_size,
                              hipStream_t stream) {
    const float* x  = (const float*)d_in[0];   // [8,4096,1024]
    const float* w1 = (const float*)d_in[1];   // [2,8,128,128]
    const float* b1 = (const float*)d_in[2];   // [2,8,128]
    const float* w2 = (const float*)d_in[3];   // [2,8,128,128]
    const float* b2 = (const float*)d_in[4];   // [2,8,128]
    float* out = (float*)d_out;

    // workspace: fragment-swizzled f16 matrices (total ~9.4 MB)
    h16* M1  = (h16*)d_ws;                     // 1024*2048
    h16* W2c = M1 + 2097152;                   // 8*256*256
    h16* Cm  = W2c + 524288;                   // 2048*1024

    afno_prep_m1<<<8192, 256, 0, stream>>>(w1, M1);
    afno_prep_w2<<<2048, 256, 0, stream>>>(w2, W2c);
    afno_prep_cm<<<8192, 256, 0, stream>>>(Cm);

    size_t lds_bytes = (size_t)(2 * MTILE * SO_STR) * sizeof(h16); // 263,168 B
    afno_main<<<32768 / MTILE, 256, lds_bytes, stream>>>(x, b1, b2, M1, W2c, Cm, out);
}
